// MultiheadLocalPosAttention_1881195676260
// MI455X (gfx1250) — matrix operand order repaired and checked by reading, not compiled, not run on hardware
//
#include <hip/hip_runtime.h>

// MultiheadLocalPosAttention for MI455X (gfx1250, wave32, WMMA).
// Pipeline (all on `stream`):
//   0) convert Wqkv / Wo / Wp2 weights fp32 -> bf16 (into ws)
//   1) qkv = x @ Wqkv^T + bqkv     (WMMA bf16 GEMM, 16x64 tile/wave, fp32 accum)
//   2) per-node pos-MLP via WMMA (B from LDS, staged by async-to-LDS copies)
//      + k/v gather + masked per-channel softmax over the 16 neighbors
//   3) out = att @ Wo^T + bo       (WMMA bf16 GEMM)
//
// Workspace layout (~134.9 MB):
//   [0, 100663296)          : qkv fp32  [32768 x 768]
//   [100663296, 134217728)  : att fp32  [32768 x 256]
//   [134217728, ...)        : bf16 Wqkv (196608), Wo (65536), Wp2 (32768)

typedef __attribute__((ext_vector_type(16))) __bf16 v16bf;
typedef __attribute__((ext_vector_type(8)))  __bf16 v8bf;
typedef __attribute__((ext_vector_type(8)))  float  v8f;

#define NPTS 32768
#define SNB  16
#define DIMC 256
#define PHID 128
#define QLD  768   // qkv row stride (3*DIM)
#define NODES_PER_BLK 8

// ---- helpers ---------------------------------------------------------------
__device__ __forceinline__ void put4(v16bf& d, int e, const float4 f) {
  d[e + 0] = (__bf16)f.x; d[e + 1] = (__bf16)f.y;
  d[e + 2] = (__bf16)f.z; d[e + 3] = (__bf16)f.w;
}
__device__ __forceinline__ void hid4(v16bf& d, int e, float dist,
                                     const float4 w, const float4 b) {
  d[e + 0] = (__bf16)fmaxf(fmaf(dist, w.x, b.x), 0.f);
  d[e + 1] = (__bf16)fmaxf(fmaf(dist, w.y, b.y), 0.f);
  d[e + 2] = (__bf16)fmaxf(fmaf(dist, w.z, b.z), 0.f);
  d[e + 3] = (__bf16)fmaxf(fmaf(dist, w.w, b.w), 0.f);
}
template <typename P>
__device__ __forceinline__ v16bf load16bf(const P* p) {   // works for global & LDS
  v8bf lo = *(const v8bf*)(p);
  v8bf hi = *(const v8bf*)(p + 16);
  return __builtin_shufflevector(lo, hi, 0,1,2,3,4,5,6,7,8,9,10,11,12,13,14,15);
}

// ---- kernel 0: weight conversion fp32 -> bf16 ------------------------------
__global__ void cvt_f32_bf16(const float* __restrict__ src,
                             __bf16* __restrict__ dst, int n) {
  int i = blockIdx.x * blockDim.x + threadIdx.x;
  if (i < n) dst[i] = (__bf16)src[i];
}

// ---- kernels 1 & 3: C[n][j] = sum_k A[n][k] * W[j][k] + bias[j] ------------
// One wave per 16x64 tile: the A fragment is converted once per K-chunk and
// feeds 4 consecutive WMMAs (4x less A traffic / cvt work than 16x16 tiles).
__global__ void gemm_nt_bf16(const float* __restrict__ A,
                             const __bf16* __restrict__ Wbf,
                             const float* __restrict__ bias,
                             float* __restrict__ C, int K, int J) {
  const int l    = threadIdx.x;
  const int half = l & 16;            // 0 or 16
  const int lm   = l & 15;
  const int rowA = blockIdx.y * 16 + lm;
  const int colBase = blockIdx.x * 64;
  v8f acc[4] = {{0.f,0.f,0.f,0.f,0.f,0.f,0.f,0.f},
                {0.f,0.f,0.f,0.f,0.f,0.f,0.f,0.f},
                {0.f,0.f,0.f,0.f,0.f,0.f,0.f,0.f},
                {0.f,0.f,0.f,0.f,0.f,0.f,0.f,0.f}};

  const int nk = K >> 5;
  for (int kk = 0; kk < nk; ++kk) {
    const int kb = kk << 5;
    // A fragment (M=row, K): lanes 0-15 hold K = kb+{0..7,16..23}; lanes 16-31 +8
    const float* ap = A + (size_t)rowA * K + kb + (half ? 8 : 0);
    v16bf af;
    put4(af, 0,  *(const float4*)(ap + 0));
    put4(af, 4,  *(const float4*)(ap + 4));
    put4(af, 8,  *(const float4*)(ap + 16));
    put4(af, 12, *(const float4*)(ap + 20));
#pragma unroll
    for (int t = 0; t < 4; ++t) {
      // B fragment (K, N=col): lanes 0-15 hold K = kb+{0..7,16..23}; lanes 16-31 +8
      const int colJ = colBase + t * 16 + lm;
      const __bf16* bp = Wbf + (size_t)colJ * K + kb + (half ? 8 : 0);
      v16bf bf = load16bf(bp);
      acc[t] = __builtin_amdgcn_wmma_f32_16x16x32_bf16(false, af, false, bf,
                                                       (short)0, acc[t], false, false);
    }
  }
#pragma unroll
  for (int t = 0; t < 4; ++t) {
    const int colJ = colBase + t * 16 + lm;
    const float b = bias[colJ];
#pragma unroll
    for (int i = 0; i < 8; ++i) {
      const int rowOut = blockIdx.y * 16 + i + (half ? 8 : 0);
      C[(size_t)rowOut * J + colJ] = acc[t][i] + b;
    }
  }
}

// ---- kernel 2: per-node pos-MLP (WMMA) + gather + masked softmax -----------
// 8 waves / block, one node per wave. The bf16 Wp2 (64 KB) is staged into LDS
// once per block with global_load_async_to_lds_b128 (ASYNCcnt), so the VMEM
// path + WGP$ stay dedicated to the gathered qkv rows. B fragments then come
// from LDS (ds_load_b128). Softmax over s: 8-wide register reduction + one
// shfl_xor(16) thanks to the C/D tile layout (channel on lanes, s on VGPRs).
__global__ void __launch_bounds__(256)
attn_pos_kernel(const float* __restrict__ qkv,
                const float* __restrict__ pos,
                const int* __restrict__ aidx,
                const unsigned char* __restrict__ mask, // jax bool = 1 byte
                const float* __restrict__ Wp1,
                const float* __restrict__ bp1,
                const __bf16* __restrict__ Wp2bf,
                const float* __restrict__ bp2,
                float* __restrict__ att) {
  __shared__ __bf16 s_wp2[DIMC * PHID];   // exactly 64 KB

  const int tid  = threadIdx.x;           // 0..255
  const int wave = tid >> 5;
  const int l    = tid & 31;
  const int half = l & 16;
  const int lm   = l & 15;
  const int n    = blockIdx.x * NODES_PER_BLK + wave;

  // ---- stage Wp2 (bf16, 65536 B) into LDS with async copies ----
  {
    const unsigned ldsBase = (unsigned)(size_t)(&s_wp2[0]); // low 32b = LDS offset
#pragma unroll
    for (int i = 0; i < 16; ++i) {
      const int byteOff = (tid + i * 256) * 16;             // 256 thr * 16 it * 16 B
      const unsigned ldsAddr = ldsBase + (unsigned)byteOff;
      const unsigned long long ga =
          (unsigned long long)(size_t)((const char*)Wp2bf + byteOff);
      asm volatile("global_load_async_to_lds_b128 %0, %1, off"
                   :: "v"(ldsAddr), "v"(ga) : "memory");
    }
    asm volatile("s_wait_asynccnt 0x0" ::: "memory");
  }
  __syncthreads();

  // ---- per-node neighbor metadata (registers + shuffles, no LDS) ----
  // every lane computes s = lm (upper half duplicates lanes 0-15)
  const int   idxv  = aidx[n * SNB + lm];
  const int   mskv  = (int)mask[n * SNB + lm];
  const float dx    = pos[idxv * 3 + 0] - pos[n * 3 + 0];
  const float dy    = pos[idxv * 3 + 1] - pos[n * 3 + 1];
  const float dz    = pos[idxv * 3 + 2] - pos[n * 3 + 2];
  const float distv = sqrtf(dx * dx + dy * dy + dz * dz);

  // A fragments: M = s = lm, K = ph; hidden = relu(dist*Wp1 + bp1)
  v16bf af[4];
#pragma unroll
  for (int kk = 0; kk < 4; ++kk) {
    const int phb = kk * 32 + (half ? 8 : 0);
    hid4(af[kk], 0,  distv, *(const float4*)(Wp1 + phb + 0),
                            *(const float4*)(bp1 + phb + 0));
    hid4(af[kk], 4,  distv, *(const float4*)(Wp1 + phb + 4),
                            *(const float4*)(bp1 + phb + 4));
    hid4(af[kk], 8,  distv, *(const float4*)(Wp1 + phb + 16),
                            *(const float4*)(bp1 + phb + 16));
    hid4(af[kk], 12, distv, *(const float4*)(Wp1 + phb + 20),
                            *(const float4*)(bp1 + phb + 20));
  }

  // this lane owns rows s = i + (half?8:0), i=0..7 of the C/D tile
  int myidx[8], mymsk[8];
#pragma unroll
  for (int i = 0; i < 8; ++i) {
    const int s = i + (half ? 8 : 0);
    myidx[i] = __shfl(idxv, s, 32);
    mymsk[i] = __shfl(mskv, s, 32);
  }

  // ---- 16 column tiles of 16 channels each ----
  for (int ct = 0; ct < 16; ++ct) {
    const int c = ct * 16 + lm;
    v8f acc = {0.f, 0.f, 0.f, 0.f, 0.f, 0.f, 0.f, 0.f};
#pragma unroll
    for (int kk = 0; kk < 4; ++kk) {
      const __bf16* bp = &s_wp2[(size_t)c * PHID + kk * 32 + (half ? 8 : 0)];
      v16bf bf = load16bf(bp);
      acc = __builtin_amdgcn_wmma_f32_16x16x32_bf16(false, af[kk], false, bf,
                                                    (short)0, acc, false, false);
    }
    const float b2 = bp2[c];
    const float qc = qkv[(size_t)n * QLD + c];

    float w8[8], vv8[8];
#pragma unroll
    for (int i = 0; i < 8; ++i) {
      const float rel = acc[i] + b2;                 // SCALE == 1
      const size_t rb = (size_t)myidx[i] * QLD + c;
      const float kc  = qkv[rb + DIMC];
      const float vc  = qkv[rb + 2 * DIMC];
      w8[i]  = mymsk[i] ? -1e30f : fmaf(kc, qc, rel);
      vv8[i] = vc + rel;
    }
    float m = w8[0];
#pragma unroll
    for (int i = 1; i < 8; ++i) m = fmaxf(m, w8[i]);
    m = fmaxf(m, __shfl_xor(m, 16, 32));
    float den = 0.f, num = 0.f;
#pragma unroll
    for (int i = 0; i < 8; ++i) {
      const float e = __expf(w8[i] - m);             // masked -> exp(-1e30) = 0
      den += e;
      num += vv8[i] * e;
    }
    den += __shfl_xor(den, 16, 32);
    num += __shfl_xor(num, 16, 32);
    if (l < 16) att[(size_t)n * DIMC + c] = num / den;
  }
}

// ---- host launcher ---------------------------------------------------------
extern "C" void kernel_launch(void* const* d_in, const int* in_sizes, int n_in,
                              void* d_out, int out_size, void* d_ws, size_t ws_size,
                              hipStream_t stream) {
  const float*         x    = (const float*)d_in[0];
  const float*         pos  = (const float*)d_in[1];
  const int*           aidx = (const int*)d_in[2];
  const unsigned char* msk  = (const unsigned char*)d_in[3];
  const float*         Wqkv = (const float*)d_in[4];
  const float*         bqkv = (const float*)d_in[5];
  const float*         Wp1  = (const float*)d_in[6];
  const float*         bp1  = (const float*)d_in[7];
  const float*         Wp2  = (const float*)d_in[8];
  const float*         bp2  = (const float*)d_in[9];
  const float*         Wo   = (const float*)d_in[10];
  const float*         bo   = (const float*)d_in[11];
  float*               out  = (float*)d_out;

  char* ws = (char*)d_ws;
  float*  qkv    = (float*)(ws + 0);
  float*  att    = (float*)(ws + (size_t)NPTS * QLD * 4);               // +100663296
  __bf16* WqkvBf = (__bf16*)(ws + (size_t)NPTS * QLD * 4
                                + (size_t)NPTS * DIMC * 4);             // +134217728
  __bf16* WoBf   = WqkvBf + (size_t)3 * DIMC * DIMC;                    // 196608 elems
  __bf16* Wp2Bf  = WoBf   + (size_t)DIMC * DIMC;                        // 65536 elems

  // 0) weight conversions (serialized on stream ahead of consumers)
  {
    int n1 = 3 * DIMC * DIMC, n2 = DIMC * DIMC, n3 = DIMC * PHID;
    cvt_f32_bf16<<<(n1 + 255) / 256, 256, 0, stream>>>(Wqkv, WqkvBf, n1);
    cvt_f32_bf16<<<(n2 + 255) / 256, 256, 0, stream>>>(Wo,   WoBf,   n2);
    cvt_f32_bf16<<<(n3 + 255) / 256, 256, 0, stream>>>(Wp2,  Wp2Bf,  n3);
  }
  // 1) qkv projection: [32768,256] x [256,768], 16x64 tile per wave
  {
    dim3 grid(QLD / 64, NPTS / 16);
    gemm_nt_bf16<<<grid, 32, 0, stream>>>(x, WqkvBf, bqkv, qkv, DIMC, QLD);
  }
  // 2) per-node pos-MLP + gather + masked per-channel softmax
  attn_pos_kernel<<<NPTS / NODES_PER_BLK, 256, 0, stream>>>(
      qkv, pos, aidx, msk, Wp1, bp1, Wp2Bf, bp2, att);
  // 3) output projection: [32768,256] x [256,256]
  {
    dim3 grid(DIMC / 64, NPTS / 16);
    gemm_nt_bf16<<<grid, 32, 0, stream>>>(att, WoBf, bo, out, DIMC, DIMC);
  }
  (void)in_sizes; (void)n_in; (void)out_size; (void)ws_size;
}
